// SpamRNN_39771397160979
// MI455X (gfx1250) — compile-verified, matrix operand order checked by reading, not run on hardware
//
#include <hip/hip_runtime.h>
#include <hip/hip_bf16.h>
#include <stdint.h>

// ---------------------------------------------------------------------------
// 2-layer LSTM (B=64, T=512, E=512, H=1024) on gfx1250 via bf16 WMMA.
//   - input projections hoisted out of the time loop as one big WMMA GEMM
//   - recurrence: one fused kernel launch per timestep (stream-ordered sync)
//   - W_hh (8 MB bf16) stays L2-resident across the 512 sequential steps
// ---------------------------------------------------------------------------

typedef __attribute__((ext_vector_type(16))) __bf16 v16bf;
typedef __attribute__((ext_vector_type(8)))  __bf16 v8bf;
typedef __attribute__((ext_vector_type(8)))  float  v8f;

#define RB  64        // batch
#define RT  512       // timesteps
#define RE  512       // embedding dim
#define RH  1024      // hidden
#define RG  4096      // 4*H (gates)
#define RBT (RB * RT) // flattened tokens = 32768
#define RBH (RB * RH)

__device__ __forceinline__ unsigned short f32_bf16(float f) {
  unsigned u = __float_as_uint(f);
  u += 0x7FFFu + ((u >> 16) & 1u);   // round-to-nearest-even
  return (unsigned short)(u >> 16);
}
__device__ __forceinline__ float sigmoidf_(float x) {
  return 1.0f / (1.0f + __expf(-x));
}

// ---------------------------------------------------------------------------
// Repack W [4H rows, K cols] (PyTorch convention: gates = x @ W^T) into the
// per-lane WMMA B-fragment layout. Fragment (nt, kt) covers output cols
// nt*16..+15, K slice kt*32..+31. Lane l holds column (l&15); halves 0..7 map
// to K offsets base..base+7 and halves 8..15 to base+16..base+23 where
// base = 8*(l>=16)  (mirror of the documented 16-bit A layout).
// ---------------------------------------------------------------------------
__global__ void k_repack(const float* __restrict__ W,
                         unsigned short* __restrict__ out, int K) {
  size_t idx = (size_t)blockIdx.x * blockDim.x + threadIdx.x;
  if (idx >= (size_t)RG * K) return;
  int h    = (int)(idx & 15);
  int lane = (int)((idx >> 4) & 31);
  size_t tile = idx >> 9;
  int nKt = K >> 5;
  int kt = (int)(tile % nKt);
  int nt = (int)(tile / nKt);
  int col  = lane & 15;
  int koff = h + ((h >= 8) ? 8 : 0) + ((lane >= 16) ? 8 : 0);
  int n = nt * 16 + col;    // row of W = gate output index
  int k = kt * 32 + koff;
  out[idx] = f32_bf16(W[(size_t)n * K + k]);
}

__global__ void k_bias(const float* __restrict__ a, const float* __restrict__ b,
                       float* __restrict__ o) {
  int i = blockIdx.x * blockDim.x + threadIdx.x;
  if (i < RG) o[i] = a[i] + b[i];
}

// Embedding gather -> bf16, X0[bt, e] row-major
__global__ void k_embed(const int* __restrict__ x, const float* __restrict__ emb,
                        unsigned short* __restrict__ X0) {
  size_t idx = (size_t)blockIdx.x * blockDim.x + threadIdx.x;
  if (idx >= (size_t)RBT * RE) return;
  int bt = (int)(idx >> 9);        // / RE (=512)
  int e  = (int)(idx & (RE - 1));
  int tok = x[bt];
  X0[idx] = f32_bf16(emb[(size_t)tok * RE + e]);
}

__global__ void k_init_state(const float* __restrict__ h0l, const float* __restrict__ c0l,
                             float* __restrict__ hS, float* __restrict__ cS,
                             unsigned short* __restrict__ hb0) {
  int i = blockIdx.x * blockDim.x + threadIdx.x;
  if (i >= RBH) return;
  float h = h0l[i];
  hS[i] = h;
  cS[i] = c0l[i];
  hb0[i] = f32_bf16(h);
}

// ---------------------------------------------------------------------------
// XP[bt, 0:4096] = Xb[bt, 0:K] @ W^T + bias   (bf16 in, fp32 out)
// One wave computes a 16-row x 64-col strip (4 N-tiles, A fragment reused).
// ---------------------------------------------------------------------------
__global__ void k_proj_gemm(const unsigned short* __restrict__ Xb,
                            const unsigned short* __restrict__ Wp,
                            const float* __restrict__ bias,
                            float* __restrict__ XP, int K) {
  const int lane = threadIdx.x & 31;
  const int wave = blockIdx.x * (blockDim.x >> 5) + (threadIdx.x >> 5);
  const int nGroups = RG / 64;        // 64 strips of 4 N-tiles
  const int mt = wave / nGroups;      // 0 .. RBT/16-1
  const int ng = wave % nGroups;
  const int row0 = mt * 16;
  const int nKt = K >> 5;
  const int l16 = lane & 15;
  const int hiK  = (lane >> 4) << 3;  // 0 or 8 : K-offset for A halves
  const int rsel = (lane >> 4) << 3;  // 0 or 8 : row offset in C layout

  v8f acc[4] = {};
  const unsigned short* arow = Xb + (size_t)(row0 + l16) * K + hiK;
  for (int kt = 0; kt < nKt; ++kt) {
    const unsigned short* ap = arow + kt * 32;
    v8bf lo = *reinterpret_cast<const v8bf*>(ap);       // K: base+0..7
    v8bf hi = *reinterpret_cast<const v8bf*>(ap + 16);  // K: base+16..23
    v16bf a = __builtin_shufflevector(lo, hi, 0,1,2,3,4,5,6,7,
                                              8,9,10,11,12,13,14,15);
#pragma unroll
    for (int j = 0; j < 4; ++j) {
      const int nt = ng * 4 + j;
      const v16bf b = *reinterpret_cast<const v16bf*>(
          Wp + (((size_t)nt * nKt + kt) * 32 + lane) * 16);
      acc[j] = __builtin_amdgcn_wmma_f32_16x16x32_bf16(
          false, a, false, b, (short)0, acc[j], false, false);
    }
  }
#pragma unroll
  for (int j = 0; j < 4; ++j) {
    const int col = (ng * 4 + j) * 16 + l16;
    const float bv = bias[col];
#pragma unroll
    for (int r = 0; r < 8; ++r) {
      const int row = row0 + r + rsel;           // C layout: M = r (+8 hi lanes)
      XP[(size_t)row * RG + col] = acc[j][r] + bv;
    }
  }
}

// ---------------------------------------------------------------------------
// One LSTM timestep, fused: gates = XP[t] + h_{t-1} @ Whh^T, then gate math.
// Block = 4 waves (M tiles over B=64), grid = 64 (hidden-column tiles).
// Each wave computes all 4 gate tiles for its (batch-tile, h-tile) so the
// elementwise update is local. h is ping-pong double-buffered (bf16).
// ---------------------------------------------------------------------------
__global__ void k_lstm_step(const unsigned short* __restrict__ hin,
                            const unsigned short* __restrict__ Whh,
                            const float* __restrict__ XP, int t,
                            float* __restrict__ cS, float* __restrict__ hS,
                            unsigned short* __restrict__ hout,
                            unsigned short* __restrict__ Hseq, int writeSeq) {
  const int lane = threadIdx.x & 31;
  const int wv   = threadIdx.x >> 5;   // 0..3 -> batch tile
  const int ntH  = blockIdx.x;         // 0..63 -> hidden col tile
  const int row0 = wv * 16;
  const int l16  = lane & 15;
  const int hiK  = (lane >> 4) << 3;
  const int rsel = (lane >> 4) << 3;
  const int nKt  = RH >> 5;            // 32

  v8f acc[4];
#pragma unroll
  for (int g = 0; g < 4; ++g) {
    const int col = g * RH + ntH * 16 + l16;   // gate order i,f,g,o
#pragma unroll
    for (int r = 0; r < 8; ++r) {
      const int b = row0 + r + rsel;
      acc[g][r] = XP[((size_t)b * RT + t) * RG + col];
    }
  }

  const unsigned short* arow = hin + (size_t)(row0 + l16) * RH + hiK;
  for (int kt = 0; kt < nKt; ++kt) {
    const unsigned short* ap = arow + kt * 32;
    v8bf lo = *reinterpret_cast<const v8bf*>(ap);
    v8bf hi = *reinterpret_cast<const v8bf*>(ap + 16);
    v16bf a = __builtin_shufflevector(lo, hi, 0,1,2,3,4,5,6,7,
                                              8,9,10,11,12,13,14,15);
#pragma unroll
    for (int g = 0; g < 4; ++g) {
      const int nt = g * 64 + ntH;    // tile index inside the 4096-wide gates
      const v16bf b = *reinterpret_cast<const v16bf*>(
          Whh + (((size_t)nt * nKt + kt) * 32 + lane) * 16);
      acc[g] = __builtin_amdgcn_wmma_f32_16x16x32_bf16(
          false, a, false, b, (short)0, acc[g], false, false);
    }
  }

#pragma unroll
  for (int r = 0; r < 8; ++r) {
    const int b   = row0 + r + rsel;
    const int col = ntH * 16 + l16;
    const size_t ci = (size_t)b * RH + col;
    const float iv = sigmoidf_(acc[0][r]);
    const float fv = sigmoidf_(acc[1][r]);
    const float gv = tanhf(acc[2][r]);
    const float ov = sigmoidf_(acc[3][r]);
    const float c  = fv * cS[ci] + iv * gv;
    cS[ci] = c;
    const float h = ov * tanhf(c);
    hS[ci] = h;
    const unsigned short hb = f32_bf16(h);
    hout[ci] = hb;
    if (writeSeq) Hseq[((size_t)b * RT + t) * RH + col] = hb;
  }
}

// sig_out[b] = sigmoid(dot(h_layer1_last[b,:], fcW) + fcb)
__global__ void k_logit(const float* __restrict__ hS1, const float* __restrict__ fcW,
                        const float* __restrict__ fcb, float* __restrict__ out) {
  __shared__ float red[128];
  const int b = blockIdx.x;
  float s = 0.f;
  for (int i = threadIdx.x; i < RH; i += blockDim.x)
    s += hS1[(size_t)b * RH + i] * fcW[i];
  red[threadIdx.x] = s;
  __syncthreads();
  for (int st = blockDim.x >> 1; st > 0; st >>= 1) {
    if ((int)threadIdx.x < st) red[threadIdx.x] += red[threadIdx.x + st];
    __syncthreads();
  }
  if (threadIdx.x == 0) out[b] = sigmoidf_(red[0] + fcb[0]);
}

// out layout: [64 sig][2*64*1024 hn][2*64*1024 cn]
__global__ void k_emit_states(const float* __restrict__ hS, const float* __restrict__ cS,
                              float* __restrict__ out) {
  int i = blockIdx.x * blockDim.x + threadIdx.x;
  const int N = 2 * RBH;
  if (i >= N) return;
  out[RB + i]     = hS[i];
  out[RB + N + i] = cS[i];
}

extern "C" void kernel_launch(void* const* d_in, const int* in_sizes, int n_in,
                              void* d_out, int out_size, void* d_ws, size_t ws_size,
                              hipStream_t stream) {
  (void)in_sizes; (void)n_in; (void)out_size; (void)ws_size;
  const int*   x     = (const int*)  d_in[0];
  const float* h0    = (const float*)d_in[1];   // [2,B,H]
  const float* c0    = (const float*)d_in[2];
  const float* emb   = (const float*)d_in[3];
  const float* Wih0  = (const float*)d_in[4];
  const float* Whh0  = (const float*)d_in[5];
  const float* bih0  = (const float*)d_in[6];
  const float* bhh0  = (const float*)d_in[7];
  const float* Wih1  = (const float*)d_in[8];
  const float* Whh1  = (const float*)d_in[9];
  const float* bih1  = (const float*)d_in[10];
  const float* bhh1  = (const float*)d_in[11];
  const float* fcW   = (const float*)d_in[12];
  const float* fcb   = (const float*)d_in[13];
  float* out = (float*)d_out;

  // ---- workspace carve-up (256B aligned) -------------------------------
  char* base = (char*)d_ws;
  size_t off = 0;
  auto carve = [&](size_t bytes) -> char* {
    off = (off + 255) & ~(size_t)255;
    char* p = base + off;
    off += bytes;
    return p;
  };
  unsigned short* wih0p = (unsigned short*)carve((size_t)RG * RE * 2);
  unsigned short* whh0p = (unsigned short*)carve((size_t)RG * RH * 2);
  unsigned short* wih1p = (unsigned short*)carve((size_t)RG * RH * 2);
  unsigned short* whh1p = (unsigned short*)carve((size_t)RG * RH * 2);
  float*          bias0 = (float*)carve((size_t)RG * 4);
  float*          bias1 = (float*)carve((size_t)RG * 4);
  unsigned short* X0    = (unsigned short*)carve((size_t)RBT * RE * 2);
  unsigned short* Hseq  = (unsigned short*)carve((size_t)RBT * RH * 2);
  float*          XP    = (float*)carve((size_t)RBT * RG * 4);   // 512 MiB, reused
  float*          hS    = (float*)carve((size_t)2 * RBH * 4);    // per-layer final h
  float*          cS    = (float*)carve((size_t)2 * RBH * 4);
  unsigned short* hb    = (unsigned short*)carve((size_t)2 * RBH * 2); // ping-pong

  // ---- one-time prep ---------------------------------------------------
  k_repack<<<(int)(((size_t)RG * RE) / 256), 256, 0, stream>>>(Wih0, wih0p, RE);
  k_repack<<<(int)(((size_t)RG * RH) / 256), 256, 0, stream>>>(Whh0, whh0p, RH);
  k_repack<<<(int)(((size_t)RG * RH) / 256), 256, 0, stream>>>(Wih1, wih1p, RH);
  k_repack<<<(int)(((size_t)RG * RH) / 256), 256, 0, stream>>>(Whh1, whh1p, RH);
  k_bias<<<(RG + 255) / 256, 256, 0, stream>>>(bih0, bhh0, bias0);
  k_bias<<<(RG + 255) / 256, 256, 0, stream>>>(bih1, bhh1, bias1);
  k_embed<<<(int)(((size_t)RBT * RE) / 256), 256, 0, stream>>>(x, emb, X0);

  const int projBlocks = (RBT / 16) * (RG / 64) / 8; // 8 waves / 256-thread block
  const int initBlocks = (RBH + 255) / 256;

  // ---- layer 0 ---------------------------------------------------------
  k_init_state<<<initBlocks, 256, 0, stream>>>(h0, c0, hS, cS, hb);
  k_proj_gemm<<<projBlocks, 256, 0, stream>>>(X0, wih0p, bias0, XP, RE);
  for (int t = 0; t < RT; ++t) {
    k_lstm_step<<<RH / 16, 128, 0, stream>>>(
        hb + (size_t)(t & 1) * RBH, whh0p, XP, t,
        cS, hS, hb + (size_t)((t + 1) & 1) * RBH, Hseq, 1);
  }

  // ---- layer 1 ---------------------------------------------------------
  k_init_state<<<initBlocks, 256, 0, stream>>>(h0 + RBH, c0 + RBH,
                                               hS + RBH, cS + RBH, hb);
  k_proj_gemm<<<projBlocks, 256, 0, stream>>>(Hseq, wih1p, bias1, XP, RH);
  for (int t = 0; t < RT; ++t) {
    k_lstm_step<<<RH / 16, 128, 0, stream>>>(
        hb + (size_t)(t & 1) * RBH, whh1p, XP, t,
        cS + RBH, hS + RBH, hb + (size_t)((t + 1) & 1) * RBH, Hseq, 0);
  }

  // ---- outputs ---------------------------------------------------------
  k_logit<<<RB, 128, 0, stream>>>(hS + RBH, fcW, fcb, out);
  k_emit_states<<<(2 * RBH + 255) / 256, 256, 0, stream>>>(hS, cS, out);
}
